// BiLSTM_CRF_29265907155260
// MI455X (gfx1250) — matrix operand order repaired and checked by reading, not compiled, not run on hardware
//
#include <hip/hip_runtime.h>
#include <hip/hip_bf16.h>

#define NUM_TAGS 32
#define START_ID 29
#define STOP_ID  30
#define PAD_ID   31
#define EMB 256
#define HID 512
#define HD  256
#define BB  64
#define TT  512
#define G4H 1024   // 4*HD

typedef __attribute__((ext_vector_type(16))) __bf16 v16bf;
typedef __attribute__((ext_vector_type(8)))  float  v8f;

static __device__ __forceinline__ unsigned short f2bf(float x){
  union{float f; unsigned u;} v; v.f = x;
  unsigned r = (v.u + 0x7FFFu + ((v.u >> 16) & 1u)) >> 16;  // RNE
  return (unsigned short)r;
}
static __device__ __forceinline__ unsigned pack_bf(float lo, float hi){
  return (unsigned)f2bf(lo) | ((unsigned)f2bf(hi) << 16);
}
static __device__ __forceinline__ float sigmoidf_(float x){ return 1.0f/(1.0f + __expf(-x)); }

static __device__ __forceinline__ v8f wmma_bf16(const unsigned* a8, const unsigned* b8, v8f c){
  union{v16bf v; unsigned u[8];} A, B;
#pragma unroll
  for(int i=0;i<8;++i){ A.u[i]=a8[i]; B.u[i]=b8[i]; }
  return __builtin_amdgcn_wmma_f32_16x16x32_bf16(false, A.v, false, B.v, (short)0, c, false, false);
}

// ISA 16-bit A-fragment (16x32) K offset for VGPR v, lane-half hi
static __device__ __forceinline__ int a_koff(int v, int hi){ return ((v<4)? 2*v : 8 + 2*v) + (hi? 8:0); }
// ISA 16-bit B-fragment (32x16) K offset
static __device__ __forceinline__ int b_koff(int v, int hi){ return ((v<4)? 2*v : 8 + 2*v) + (hi? 8:0); }

// ---------------------------------------------------------------- convert
__global__ void k_f32_to_bf16(unsigned short* dst, const float* src, int n){
  int i = blockIdx.x*blockDim.x + threadIdx.x;
  if(i<n) dst[i] = f2bf(src[i]);
}

// ---------------------------------------------------- embedding + x@Wih^T + b
// grid: 2048 M-tiles of the flattened [T*B, EMB] activation. block = 256 (8 waves).
// Embedding rows are staged into LDS with the gfx1250 async global->LDS path
// (ASYNCcnt). The staging buffer is then read through a VOLATILE pointer when
// packing the bf16 A fragments: the asm writer is invisible to clang's AA, and
// non-volatile reads of a "never-written" LDS array get folded/CSE'd (observed
// in round 2: all 8 chunk WMMAs shared one A octet). Volatile forces real,
// distinct ds loads for each of the 64 fragment registers.
__global__ void __launch_bounds__(256)
k_embed_proj(const int* inp, const float* emb,
             const unsigned short* wih_f, const unsigned short* wih_b,
             const float* b_f, const float* b_b,
             float* xw_f, float* xw_b){
  __shared__ float lds_x[16*EMB];            // 16 rows x 256 f32 = 16 KB (LDS offset 0)
  int mt = blockIdx.x;
  int tid = threadIdx.x;
  { // each thread async-copies 64B of one embedding row into LDS
    int r  = tid >> 4;
    int kb = (tid & 15) * 16;
    int R  = mt*16 + r;
    int t  = R >> 6, b = R & 63;
    int tok = inp[b*TT + t];
    unsigned ldsoff = (unsigned)((r*EMB + kb) * 4);              // lds_x is at LDS offset 0
    unsigned goff   = (unsigned)(((unsigned)tok*EMB + kb) * 4);
    // GVS mode: INST_OFFSET advances both the LDS and global addresses.
    asm volatile(
      "global_load_async_to_lds_b128 %0, %1, %2 offset:0\n\t"
      "global_load_async_to_lds_b128 %0, %1, %2 offset:16\n\t"
      "global_load_async_to_lds_b128 %0, %1, %2 offset:32\n\t"
      "global_load_async_to_lds_b128 %0, %1, %2 offset:48\n\t"
      "s_wait_asynccnt 0x0"
      :: "v"(ldsoff), "v"(goff), "s"(emb) : "memory");
  }
  __syncthreads();
  int w = tid >> 5, lane = tid & 31;
  int hi = lane >> 4, ln = lane & 15, h8 = hi*8;
  // hoist all 8 A K-chunks into registers (pack f32->bf16); reused over 16 N-tiles
  volatile const float* lx = lds_x;          // defeat undef-folding of async-written LDS
  unsigned Areg[8][8];
#pragma unroll
  for(int kc=0;kc<8;++kc)
#pragma unroll
    for(int v=0;v<8;++v){
      int k = kc*32 + a_koff(v,hi);
      float x0 = lx[ln*EMB + k];
      float x1 = lx[ln*EMB + k + 1];
      Areg[kc][v] = pack_bf(x0, x1);
    }
  for(int i=0;i<16;++i){
    int nt = w*16 + i;                         // 0..127 = {dir x 64 N-tiles}
    int dir = nt >> 6, ntl = nt & 63;
    const unsigned short* W = dir ? wih_b : wih_f;   // [1024][256] bf16, B[k][n]=W[n][k]
    const float* bias = dir ? b_b : b_f;
    float* xw = dir ? xw_b : xw_f;
    int n = ntl*16 + ln;
    const unsigned short* Wn = W + n*EMB;
    // double-buffered B fragments: overlap next chunk's loads with current WMMA
    unsigned Br[2][8];
#pragma unroll
    for(int v=0;v<8;++v) Br[0][v] = *(const unsigned*)&Wn[b_koff(v,hi)];
    v8f acc = {};
#pragma unroll
    for(int kc=0;kc<8;++kc){
      int cur = kc & 1, nxt = cur ^ 1;
      if(kc < 7){
#pragma unroll
        for(int v=0;v<8;++v) Br[nxt][v] = *(const unsigned*)&Wn[(kc+1)*32 + b_koff(v,hi)];
      }
      acc = wmma_bf16(Areg[kc], Br[cur], acc);
    }
    float bv = bias[n];
#pragma unroll
    for(int r=0;r<8;++r){
      int row = mt*16 + h8 + r;
      xw[row*G4H + n] = acc[r] + bv;
    }
  }
}

// ------------------------------------------------------------------ LSTM scan
// grid = 2 (dir). 1024 threads = 32 waves; h kept as bf16 double buffer in LDS,
// c/h f32 state in registers. Wave owns (1 hid-tile x all 4 gates) x (2 m-tiles):
// every Whh B-fragment is reused for two WMMAs, halving the per-step L2 stream
// on the serial critical path. 64 WMMAs / wave / step, one barrier / step.
__global__ void __launch_bounds__(1024)
k_lstm(const int* mask, const unsigned short* whh_all, const float* xw_f, const float* xw_b,
       unsigned short* hs_f, unsigned short* hs_b){
  extern __shared__ unsigned short hbuf[];     // 2 x [64][256] bf16 = 64 KB
  unsigned short* h0 = hbuf;
  unsigned short* h1 = hbuf + BB*HD;
  int dir = blockIdx.x;
  const unsigned short* whh = whh_all + dir*(G4H*HD);   // [1024][256] bf16
  const float* xw = dir ? xw_b : xw_f;
  unsigned short* hs = dir ? hs_b : hs_f;
  int tid = threadIdx.x;
  for(int i=tid;i<BB*HD;i+=1024) h0[i]=0;
  __syncthreads();
  int w = tid>>5, lane = tid&31;
  int j   = w & 15;            // hid-tile 0..15
  int mtp = w >> 4;            // m-tile pair: handles m-tiles {2*mtp, 2*mtp+1}
  int hi = lane>>4, ln = lane&15, h8 = hi*8;
  int hid = j*16 + ln;
  float creg[2][8], hreg[2][8];
#pragma unroll
  for(int mi=0;mi<2;++mi)
#pragma unroll
    for(int r=0;r<8;++r){ creg[mi][r]=0.f; hreg[mi][r]=0.f; }

  for(int s=0;s<TT;++s){
    int t = dir ? (TT-1-s) : s;
    const unsigned short* hin = (s&1)? h1 : h0;
    unsigned short*       hout= (s&1)? h0 : h1;
    float mrow[2][8];
#pragma unroll
    for(int mi=0;mi<2;++mi)
#pragma unroll
      for(int r=0;r<8;++r)
        mrow[mi][r] = (float)mask[((mtp*2+mi)*16 + h8 + r)*TT + t];
    const float* xwt = xw + (size_t)(t*BB)*G4H;
    float G[4][2][8];
#pragma unroll
    for(int g=0; g<4; ++g){
      int n = g*HD + hid;                      // 4H output column
      const unsigned short* Wn = whh + n*HD;
      v8f acc0 = {}, acc1 = {};
      unsigned Br[2][8];
#pragma unroll
      for(int v=0;v<8;++v) Br[0][v] = *(const unsigned*)&Wn[b_koff(v,hi)];
#pragma unroll
      for(int kc=0;kc<8;++kc){
        int cur = kc & 1, nxt = cur ^ 1;
        if(kc < 7){
#pragma unroll
          for(int v=0;v<8;++v) Br[nxt][v] = *(const unsigned*)&Wn[(kc+1)*32 + b_koff(v,hi)];
        }
        unsigned Ar[8];
#pragma unroll
        for(int v=0;v<8;++v)
          Ar[v] = *(const unsigned*)&hin[((mtp*2+0)*16 + ln)*HD + kc*32 + a_koff(v,hi)];
        acc0 = wmma_bf16(Ar, Br[cur], acc0);
#pragma unroll
        for(int v=0;v<8;++v)
          Ar[v] = *(const unsigned*)&hin[((mtp*2+1)*16 + ln)*HD + kc*32 + a_koff(v,hi)];
        acc1 = wmma_bf16(Ar, Br[cur], acc1);
      }
#pragma unroll
      for(int r=0;r<8;++r){
        G[g][0][r] = acc0[r] + xwt[((mtp*2+0)*16 + h8 + r)*G4H + n];
        G[g][1][r] = acc1[r] + xwt[((mtp*2+1)*16 + h8 + r)*G4H + n];
      }
    }
#pragma unroll
    for(int mi=0;mi<2;++mi){
#pragma unroll
      for(int r=0;r<8;++r){
        float ig = sigmoidf_(G[0][mi][r]);
        float fg = sigmoidf_(G[1][mi][r]);
        float gg = tanhf(G[2][mi][r]);
        float og = sigmoidf_(G[3][mi][r]);
        float cn = fg*creg[mi][r] + ig*gg;
        float hn = og*tanhf(cn);
        float m  = mrow[mi][r];
        creg[mi][r] = m*cn + (1.f-m)*creg[mi][r];
        float h = m*hn + (1.f-m)*hreg[mi][r];
        hreg[mi][r] = h;
        unsigned short hb = f2bf(h);
        int row = (mtp*2+mi)*16 + h8 + r;
        hout[row*HD + hid] = hb;
        hs[(size_t)(t*BB + row)*HD + hid] = hb;
      }
    }
    __syncthreads();
  }
}

// ------------------------------------------------------------- tag projection
// emit[t][b][tag] = ([h_f;h_b] @ Wtag^T + b_tag) * mask. M=32768, K=512, N=32.
__global__ void __launch_bounds__(128)
k_tagproj(const int* mask, const unsigned short* hs_f, const unsigned short* hs_b,
          const unsigned short* wtag, const float* b_tag, float* emit){
  int w = threadIdx.x>>5, lane = threadIdx.x&31;
  int mt = blockIdx.x*4 + w;
  int hi = lane>>4, ln = lane&15, h8 = hi*8;
  int R0 = mt*16;
#pragma unroll
  for(int nt=0;nt<2;++nt){
    int n = nt*16 + ln;
    v8f acc = {};
#pragma unroll
    for(int kc=0;kc<16;++kc){
      const unsigned short* A = (kc<8)? hs_f : hs_b;   // rows of [T*B][256] bf16
      int kl = (kc&7)*32;
      unsigned Ar[8], Br[8];
#pragma unroll
      for(int v=0;v<8;++v){
        Ar[v] = *(const unsigned*)&A[(size_t)(R0+ln)*HD + kl + a_koff(v,hi)];
        Br[v] = *(const unsigned*)&wtag[n*HID + kc*32 + b_koff(v,hi)];
      }
      acc = wmma_bf16(Ar, Br, acc);
    }
    float bv = b_tag[n];
#pragma unroll
    for(int r=0;r<8;++r){
      int row = R0 + h8 + r;
      int t = row>>6, b = row&63;
      float m = (float)mask[b*TT + t];
      emit[(size_t)row*NUM_TAGS + n] = (acc[r] + bv) * m;
    }
  }
}

// ---------------------------------------------------------------- gold score
__global__ void __launch_bounds__(64)
k_gold(const int* gold, const int* mask, const float* emit, const float* trans, float* gscore){
  int b = threadIdx.x;
  if(b >= BB) return;
  int len = 0;
  for(int t=0;t<TT;++t) len += mask[b*TT + t];
  float gs = 0.f;
  for(int i=0;i<TT-1;++i){
    if(mask[b*TT + i]){
      int gn = gold[b*TT + i + 1];
      int gc = gold[b*TT + i];
      gs += emit[(size_t)(i*BB + b)*NUM_TAGS + gn] + trans[gn*NUM_TAGS + gc];
    }
  }
  int last = gold[b*TT + (len-1)];
  gs += trans[STOP_ID*NUM_TAGS + last];
  gscore[b] = gs;
}

// -------------------------------------------------------- CRF forward (Z - g)
// One wave per batch row: lane == tag (NUM_TAGS == wave32). Score in registers,
// cross-tag logsumexp via __shfl. trans row cached in 32 VGPRs.
__global__ void __launch_bounds__(256)
k_crf(const int* mask, const float* emit, const float* trans, const float* gscore, float* out){
  int w = threadIdx.x>>5, lane = threadIdx.x&31;   // lane = destination tag j
  int b = blockIdx.x*8 + w;
  float tr[NUM_TAGS];
#pragma unroll
  for(int k=0;k<NUM_TAGS;++k) tr[k] = trans[lane*NUM_TAGS + k];
  float tstop = trans[STOP_ID*NUM_TAGS + lane];
  float score = (lane == STOP_ID) ? 0.f : -10000.f;
  for(int t=0;t<TT;++t){
    if(mask[b*TT + t]){                            // uniform across the wave
      float e = emit[(size_t)(t*BB + b)*NUM_TAGS + lane];
      float vk[NUM_TAGS];
      float mx = -3.4e38f;
#pragma unroll
      for(int k=0;k<NUM_TAGS;++k){
        float sk = __shfl(score, k, 32);
        vk[k] = sk + tr[k];
        mx = fmaxf(mx, vk[k]);
      }
      float ssum = 0.f;
#pragma unroll
      for(int k=0;k<NUM_TAGS;++k) ssum += __expf(vk[k] - mx);
      score = e + mx + __logf(ssum);
    }
  }
  float f = score + tstop;
  float mx = f;
#pragma unroll
  for(int off=16; off>0; off>>=1) mx = fmaxf(mx, __shfl_xor(mx, off, 32));
  float s = __expf(f - mx);
#pragma unroll
  for(int off=16; off>0; off>>=1) s += __shfl_xor(s, off, 32);
  if(lane==0) out[b] = (mx + __logf(s)) - gscore[b];
}

// ---------------------------------------------------------------------- host
extern "C" void kernel_launch(void* const* d_in, const int* in_sizes, int n_in,
                              void* d_out, int out_size, void* d_ws, size_t ws_size,
                              hipStream_t stream){
  const int*   inp   = (const int*)  d_in[0];
  const int*   gold  = (const int*)  d_in[1];
  const int*   mask  = (const int*)  d_in[2];
  const float* emb   = (const float*)d_in[3];
  const float* Wih_f = (const float*)d_in[4];
  const float* Whh_f = (const float*)d_in[5];
  const float* b_f   = (const float*)d_in[6];
  const float* Wih_b = (const float*)d_in[7];
  const float* Whh_b = (const float*)d_in[8];
  const float* b_b   = (const float*)d_in[9];
  const float* W_tag = (const float*)d_in[10];
  const float* b_tag = (const float*)d_in[11];
  const float* trans = (const float*)d_in[12];
  float* out = (float*)d_out;
  (void)in_sizes; (void)n_in; (void)out_size; (void)ws_size;

  char* ws = (char*)d_ws;
  size_t off = 0;
  auto alloc = [&](size_t bytes)->char*{
    char* p = ws + off; off += (bytes + 255) & ~(size_t)255; return p;
  };
  const int ncv = G4H*EMB;                                     // 262144
  unsigned short* wih_bf  = (unsigned short*)alloc((size_t)2*ncv*2);
  unsigned short* whh_bf  = (unsigned short*)alloc((size_t)2*ncv*2);
  unsigned short* wtag_bf = (unsigned short*)alloc((size_t)NUM_TAGS*HID*2);
  float*          gscore  = (float*)         alloc((size_t)BB*4);
  float*          xw_f    = (float*)         alloc((size_t)TT*BB*G4H*4);
  float*          xw_b    = (float*)         alloc((size_t)TT*BB*G4H*4);
  unsigned short* hs_f    = (unsigned short*)alloc((size_t)TT*BB*HD*2);
  unsigned short* hs_b    = (unsigned short*)alloc((size_t)TT*BB*HD*2);
  float*          emit    = (float*)         alloc((size_t)TT*BB*NUM_TAGS*4);

  k_f32_to_bf16<<<(ncv+255)/256,256,0,stream>>>(wih_bf,       Wih_f, ncv);
  k_f32_to_bf16<<<(ncv+255)/256,256,0,stream>>>(wih_bf + ncv, Wih_b, ncv);
  k_f32_to_bf16<<<(ncv+255)/256,256,0,stream>>>(whh_bf,       Whh_f, ncv);
  k_f32_to_bf16<<<(ncv+255)/256,256,0,stream>>>(whh_bf + ncv, Whh_b, ncv);
  k_f32_to_bf16<<<(NUM_TAGS*HID+255)/256,256,0,stream>>>(wtag_bf, W_tag, NUM_TAGS*HID);

  k_embed_proj<<<(TT*BB)/16, 256, 0, stream>>>(inp, emb, wih_bf, wih_bf+ncv,
                                               b_f, b_b, xw_f, xw_b);
  k_lstm<<<2, 1024, 2*BB*HD*sizeof(unsigned short), stream>>>(mask, whh_bf,
                                                              xw_f, xw_b, hs_f, hs_b);
  k_tagproj<<<(TT*BB)/16/4, 128, 0, stream>>>(mask, hs_f, hs_b, wtag_bf, b_tag, emit);
  k_gold<<<1, 64, 0, stream>>>(gold, mask, emit, trans, gscore);
  k_crf<<<8, 256, 0, stream>>>(mask, emit, trans, gscore, out);
}
